// SimpleSelectiveSSMBlock_60138132078598
// MI455X (gfx1250) — matrix operand order repaired and checked by reading, not compiled, not run on hardware
//
#include <hip/hip_runtime.h>
#include <stdint.h>

#define D_MODEL 1024
#define D_STATE 16
#define BB 4
#define TT 2048
#define MROWS (BB * TT) /* 8192 */

#define LDSTR 40   /* padded LDS row stride in halfs: 80B, 16B aligned, bank-spread */

typedef _Float16 v16h __attribute__((ext_vector_type(16)));
typedef _Float16 v8h  __attribute__((ext_vector_type(8)));
typedef float    v8f  __attribute__((ext_vector_type(8)));
typedef int      v4i  __attribute__((ext_vector_type(4)));

#define __AS(n) __attribute__((address_space(n)))
typedef __AS(1) v4i* async_gptr_t;   // global source, int4-typed per builtin signature
typedef __AS(3) v4i* async_lptr_t;   // LDS destination, int4-typed

#if __has_builtin(__builtin_amdgcn_global_load_async_to_lds_b128) && \
    __has_builtin(__builtin_amdgcn_s_wait_asynccnt)
#define HAVE_ASYNC_LDS 1
#else
#define HAVE_ASYNC_LDS 0
#endif

// ---------------------------------------------------------------------------
// f32 -> f16 conversion (for WMMA operands)
// ---------------------------------------------------------------------------
__global__ __launch_bounds__(256) void f32_to_f16_kernel(const float* __restrict__ in,
                                                         _Float16* __restrict__ out, int n) {
    int i = blockIdx.x * blockDim.x + threadIdx.x;
    if (i < n) out[i] = (_Float16)in[i];
}

// ---------------------------------------------------------------------------
// Stage one 128x32 f16 K-slice of A and of B into LDS.
// 512 16B-chunks per tile; 256 threads -> 2 chunks each per tile (4 async/thread).
// ---------------------------------------------------------------------------
__device__ __forceinline__ void stage_async(const _Float16* gA, const _Float16* gB,
                                            _Float16 (*As)[LDSTR], _Float16 (*Bs)[LDSTR],
                                            int tid, int K) {
#if HAVE_ASYNC_LDS
#pragma unroll
    for (int i = 0; i < 2; ++i) {
        const int chunk = tid + i * 256;
        const int row = chunk >> 2;
        const int c   = (chunk & 3) * 8;
        __builtin_amdgcn_global_load_async_to_lds_b128(
            (async_gptr_t)(uintptr_t)(gA + (size_t)row * K + c),
            (async_lptr_t)(uintptr_t)&As[row][c], 0, 0);
        __builtin_amdgcn_global_load_async_to_lds_b128(
            (async_gptr_t)(uintptr_t)(gB + (size_t)row * K + c),
            (async_lptr_t)(uintptr_t)&Bs[row][c], 0, 0);
    }
#endif
}

__device__ __forceinline__ void stage_sync(const _Float16* gA, const _Float16* gB,
                                           _Float16 (*As)[LDSTR], _Float16 (*Bs)[LDSTR],
                                           int tid, int K) {
    __syncthreads();   // previous compute finished before overwrite
#pragma unroll
    for (int i = 0; i < 2; ++i) {
        const int chunk = tid + i * 256;
        const int row = chunk >> 2;
        const int c   = (chunk & 3) * 8;
        v8h a = *(const v8h*)(gA + (size_t)row * K + c);
        v8h b = *(const v8h*)(gB + (size_t)row * K + c);
        *(v8h*)&As[row][c] = a;
        *(v8h*)&Bs[row][c] = b;
    }
    __syncthreads();
}

// ---------------------------------------------------------------------------
// One K=32 step for a 32x64 wave tile: fragments from LDS, 8 WMMA.
// Fragment k order: contiguous, k = 16*halfSel + s, same for A 16x32 and B 32x16.
// ---------------------------------------------------------------------------
__device__ __forceinline__ void compute_stage(const _Float16 (*As)[LDSTR],
                                              const _Float16 (*Bs)[LDSTR],
                                              int mOff, int nOff, int l16, int halfSel,
                                              v8f acc[2][4]) {
    v16h bfrag[4];
#pragma unroll
    for (int nsub = 0; nsub < 4; ++nsub) {
        const _Float16* bp = &Bs[nOff + nsub * 16 + l16][16 * halfSel];
        v8h b0 = *(const v8h*)bp;          // ds_load_b128
        v8h b1 = *(const v8h*)(bp + 8);    // ds_load_b128
#pragma unroll
        for (int i = 0; i < 8; ++i) { bfrag[nsub][i] = b0[i]; bfrag[nsub][8 + i] = b1[i]; }
    }
#pragma unroll
    for (int msub = 0; msub < 2; ++msub) {
        const _Float16* ap = &As[mOff + msub * 16 + l16][0];
        v8h a0 = *(const v8h*)(ap + 16 * halfSel);       // K 0..7 / 16..23
        v8h a1 = *(const v8h*)(ap + 16 * halfSel + 8);   // K 8..15 / 24..31
        v16h afrag;
#pragma unroll
        for (int i = 0; i < 8; ++i) { afrag[i] = a0[i]; afrag[8 + i] = a1[i]; }
#pragma unroll
        for (int nsub = 0; nsub < 4; ++nsub)
            acc[msub][nsub] = __builtin_amdgcn_wmma_f32_16x16x32_f16(
                false, afrag, false, bfrag[nsub], (short)0, acc[msub][nsub], false, false);
    }
}

// ---------------------------------------------------------------------------
// C[M,N] = A[M,K] * W[N,K]^T + bias (opt. softplus). f16 in / f32 out.
// 128x128 block tile staged through LDS, double-buffered async DMA.
// ---------------------------------------------------------------------------
template <bool SOFTPLUS>
__global__ __launch_bounds__(256) void gemm_wmma_lds_kernel(
    const _Float16* __restrict__ A,    // [M, K]
    const _Float16* __restrict__ W,    // [N, K]
    const float*    __restrict__ bias, // [N]
    float*          __restrict__ C,    // [M, N]
    int M, int N, int K)
{
    __shared__ __align__(16) _Float16 As[2][128][LDSTR];
    __shared__ __align__(16) _Float16 Bs[2][128][LDSTR];

    const int tid  = threadIdx.x;
    const int wave = tid >> 5;
    const int lane = tid & 31;
    const int l16  = lane & 15;
    const int halfSel = lane >> 4;

    const int nBlocks = N / 128;
    const int mBlock  = blockIdx.x / nBlocks;
    const int nBlock  = blockIdx.x % nBlocks;

    const int mOff = (wave & 3) * 32;   // 4 waves along M
    const int nOff = (wave >> 2) * 64;  // 2 waves along N

    const _Float16* gA = A + (size_t)mBlock * 128 * K;
    const _Float16* gB = W + (size_t)nBlock * 128 * K;

    v8f acc[2][4] = {};

    const int nStages = K / 32;
#if HAVE_ASYNC_LDS
    stage_async(gA, gB, As[0], Bs[0], tid, K);
    for (int it = 0; it < nStages - 1; ++it) {
        stage_async(gA + (it + 1) * 32, gB + (it + 1) * 32,
                    As[(it + 1) & 1], Bs[(it + 1) & 1], tid, K);
        __builtin_amdgcn_s_wait_asynccnt(4);   // current stage done, next in flight
        __syncthreads();
        compute_stage(As[it & 1], Bs[it & 1], mOff, nOff, l16, halfSel, acc);
        __syncthreads();
    }
    __builtin_amdgcn_s_wait_asynccnt(0);
    __syncthreads();
    compute_stage(As[(nStages - 1) & 1], Bs[(nStages - 1) & 1], mOff, nOff, l16, halfSel, acc);
#else
    for (int it = 0; it < nStages; ++it) {
        stage_sync(gA + it * 32, gB + it * 32, As[0], Bs[0], tid, K);
        compute_stage(As[0], Bs[0], mOff, nOff, l16, halfSel, acc);
    }
#endif

    // Epilogue: C VGPR i -> row = base + i + 8*halfSel, col = base + l16
#pragma unroll
    for (int msub = 0; msub < 2; ++msub) {
#pragma unroll
        for (int nsub = 0; nsub < 4; ++nsub) {
            const int col = nBlock * 128 + nOff + nsub * 16 + l16;
            const float bv = bias[col];
#pragma unroll
            for (int i = 0; i < 8; ++i) {
                const int row = mBlock * 128 + mOff + msub * 16 + i + 8 * halfSel;
                float v = acc[msub][nsub][i] + bv;
                if (SOFTPLUS) v = (v > 20.f) ? v : log1pf(__expf(v));
                C[(size_t)row * N + col] = v;
            }
        }
    }
}

// ---------------------------------------------------------------------------
// Sequential selective-state scan: one thread per (b,d) row, 16 FP32 states
// in registers, y written as f16 for the output GEMM.
// ---------------------------------------------------------------------------
__global__ __launch_bounds__(256) void ssm_scan_kernel(
    const float* __restrict__ x,        // [B,T,D] f32
    const float* __restrict__ delta,    // [B,T,D] f32
    const float* __restrict__ a_log,    // [D, 16]
    const float* __restrict__ b_param,  // [D, 16]
    _Float16*    __restrict__ y16)      // [B,T,D] f16
{
    const int tid = blockIdx.x * blockDim.x + threadIdx.x;
    if (tid >= BB * D_MODEL) return;
    const int b = tid / D_MODEL;
    const int d = tid % D_MODEL;

    float apos[D_STATE], bp[D_STATE], st[D_STATE];
#pragma unroll
    for (int s = 0; s < D_STATE; ++s) {
        const float al = a_log[d * D_STATE + s];
        apos[s] = ((al > 20.f) ? al : log1pf(__expf(al))) + 1e-4f;
        bp[s]   = b_param[d * D_STATE + s];
        st[s]   = 0.f;
    }

    const float* xp = x     + (size_t)b * TT * D_MODEL + d;
    const float* dp = delta + (size_t)b * TT * D_MODEL + d;
    _Float16*    yp = y16   + (size_t)b * TT * D_MODEL + d;

    for (int t = 0; t < TT; ++t) {
        const float dt  = dp[(size_t)t * D_MODEL];
        const float xv  = xp[(size_t)t * D_MODEL];
        const float dbx = dt * xv;
        float ysum = 0.f;
#pragma unroll
        for (int s = 0; s < D_STATE; ++s) {
            const float ab = __expf(-dt * apos[s]);   // v_exp_f32, 16 independent
            st[s] = ab * st[s] + dbx * bp[s];
            ysum += st[s];
        }
        yp[(size_t)t * D_MODEL] = (_Float16)ysum;
    }
}

// ---------------------------------------------------------------------------
// Workspace layout (bytes):
//   x16   @ 0          : 16,777,216
//   Wd16  @ 16,777,216 :  2,097,152
//   Wo16  @ 18,874,368 :  2,097,152
//   y16   @ 20,971,520 : 16,777,216
//   delta @ 37,748,736 : 33,554,432   (total 71,303,168)
// ---------------------------------------------------------------------------
extern "C" void kernel_launch(void* const* d_in, const int* in_sizes, int n_in,
                              void* d_out, int out_size, void* d_ws, size_t ws_size,
                              hipStream_t stream) {
    const float* x       = (const float*)d_in[0];
    const float* Wd      = (const float*)d_in[1];
    const float* bd      = (const float*)d_in[2];
    const float* a_log   = (const float*)d_in[3];
    const float* b_param = (const float*)d_in[4];
    const float* Wo      = (const float*)d_in[5];
    const float* bo      = (const float*)d_in[6];
    float* out           = (float*)d_out;

    char* ws = (char*)d_ws;
    _Float16* x16   = (_Float16*)(ws + 0);
    _Float16* Wd16  = (_Float16*)(ws + 16777216);
    _Float16* Wo16  = (_Float16*)(ws + 18874368);
    _Float16* y16   = (_Float16*)(ws + 20971520);
    float*    delta = (float*)   (ws + 37748736);

    const int nX = MROWS * D_MODEL;
    const int nW = D_MODEL * D_MODEL;

    f32_to_f16_kernel<<<(nX + 255) / 256, 256, 0, stream>>>(x,  x16,  nX);
    f32_to_f16_kernel<<<(nW + 255) / 256, 256, 0, stream>>>(Wd, Wd16, nW);
    f32_to_f16_kernel<<<(nW + 255) / 256, 256, 0, stream>>>(Wo, Wo16, nW);

    // delta = softplus(x * Wd^T + bd): (8192/128)*(1024/128) = 512 blocks
    gemm_wmma_lds_kernel<true><<<512, 256, 0, stream>>>(x16, Wd16, bd, delta,
                                                        MROWS, D_MODEL, D_MODEL);

    // selective scan: 4096 rows, one thread each
    ssm_scan_kernel<<<(BB * D_MODEL) / 256, 256, 0, stream>>>(x, delta, a_log,
                                                              b_param, y16);

    // out = y * Wo^T + bo
    gemm_wmma_lds_kernel<false><<<512, 256, 0, stream>>>(y16, Wo16, bo, out,
                                                         MROWS, D_MODEL, D_MODEL);
}
